// FIA_61306363183245
// MI455X (gfx1250) — matrix operand order repaired and checked by reading, not compiled, not run on hardware
//
#include <hip/hip_runtime.h>
#include <hip/hip_bf16.h>
#include <math.h>
#include <stdint.h>

// ---------------------------------------------------------------------------
// MI455X (gfx1250) implementation: dynamic-cluster attention.
// Dense math: v_wmma_f32_16x16x32_bf16 (wave32 WMMA).
// B-panel staging: Tensor Data Mover (tensor_load_to_lds) -> LDS, double
// buffered, synchronized with s_wait_tensorcnt + workgroup barrier.
// Full K-chunks run a fully unrolled 16-WMMA body with constant LDS strides.
// ---------------------------------------------------------------------------

typedef __bf16 bf16_t;
typedef __attribute__((ext_vector_type(16))) __bf16 v16bf;
typedef __attribute__((ext_vector_type(8)))  __bf16 v8bf;
typedef __attribute__((ext_vector_type(8)))  float  v8f;
typedef __attribute__((ext_vector_type(4)))  unsigned int v4u;
typedef __attribute__((ext_vector_type(8)))  int v8i;
typedef __attribute__((ext_vector_type(4)))  int v4i;

#define BB 4
#define NN 9216
#define CC 384
#define NR3 3072          // N/3 region size
#define HH 8
#define DD 48             // head dim
#define DP 64             // padded head dim (multiple of 32 for WMMA K)
#define EE 4
#define MAXAGG 320
#define MP 960            // padded merged-token count (3*320), mult of 64
#define SORT_S 16384
#define TOPK_S 4096
#define KC 128            // K-chunk staged in LDS per TDM transfer
#define FINF __builtin_inff()

// ===========================================================================
// TDM: issue a 2D tensor_load_to_lds of a [rows x chunk] bf16 tile (row
// stride ldb elements) into LDS at byte offset lds_off. Rows beyond
// tensor_dim1 are zero-filled by hardware (OOB reads return zero).
// Descriptor layout per CDNA5 ISA section 8.3/8.4.
// ===========================================================================
__device__ __forceinline__ void tdm_load_tile(const bf16_t* gsrc, unsigned lds_off,
                                              int chunk, int rows, long ldb)
{
    unsigned long long ga = (unsigned long long)(uintptr_t)gsrc;
    v4u g0;
    g0.x = 1u;                                               // count=1 (valid), user mode
    g0.y = lds_off;                                          // lds_addr (bytes)
    g0.z = (unsigned)(ga & 0xFFFFFFFFu);                     // global_addr[31:0]
    g0.w = (unsigned)((ga >> 32) & 0x1FFFFFFu) | (2u << 30); // global_addr[56:32] | type=2
    v8i g1;
    g1[0] = (int)(1u << 16);                                 // data_size=1 (2 bytes)
    g1[1] = (int)((unsigned)(chunk & 0xFFFF) << 16);         // tensor_dim0[15:0]
    g1[2] = (int)((unsigned)(rows  & 0xFFFF) << 16);         // tensor_dim0[31:16]=0 | tensor_dim1[15:0]
    g1[3] = (int)((unsigned)(chunk & 0xFFFF) << 16);         // tensor_dim1[31:16]=0 | tile_dim0
    g1[4] = 64;                                              // tile_dim1=64, tile_dim2=0
    g1[5] = (int)(unsigned)(ldb & 0xFFFFFFFFl);              // tensor_dim0_stride[31:0]
    g1[6] = 0;                                               // stride hi / dim1_stride lo
    g1[7] = 0;
    v4i z4 = { 0, 0, 0, 0 };
#if __clang_major__ >= 23
    v8i z8 = { 0, 0, 0, 0, 0, 0, 0, 0 };
    __builtin_amdgcn_tensor_load_to_lds(g0, g1, z4, z4, z8, 0);
#else
    __builtin_amdgcn_tensor_load_to_lds(g0, g1, z4, z4, 0);
#endif
}

// ===========================================================================
// Generic WMMA GEMM:  C[M x N] = accum?C:0 + alpha_eff * A[M x K] * B[N x K]^T (+bias)
// A,B bf16 row-major; C fp32 row-major. K must be a multiple of 32.
// Block = 8 waves = 128(M) x 64(N); each wave computes a 16x64 tile.
// B panel (64 rows x KC) is DMA'd into LDS by the TDM (double buffered);
// A fragments are software-pipelined global b128 loads.
// Fragment k order (A 16x32, B 32x16): contiguous, lane half h holds k = 16h..16h+15.
// ===========================================================================
__global__ void __launch_bounds__(256)
k_wmma_gemm_nt(const bf16_t* __restrict__ A, long lda,
               const bf16_t* __restrict__ B, long ldb,
               float* __restrict__ C, long ldc,
               int M, int N, int K,
               float alpha,
               const float* __restrict__ alpha_vec,
               const int*   __restrict__ alpha_sel,
               const int*   __restrict__ b_sel, long b_stride,
               const float* __restrict__ bias,
               int accum)
{
    __shared__ __align__(64) bf16_t Bs[2][64 * KC];

    if (b_sel) B += (long)b_sel[0] * b_stride;
    float aeff = alpha;
    if (alpha_vec) aeff *= alpha_vec[alpha_sel ? alpha_sel[0] : 0];

    const int wave = threadIdx.x >> 5;
    const int lane = threadIdx.x & 31;
    const int half = lane >> 4;     // lane group 0/1
    const int lr   = lane & 15;

    const int m0 = blockIdx.y * 128 + wave * 16;
    const int n0 = blockIdx.x * 64;
    const int rows_rem = (N - n0) > 64 ? 64 : (N - n0);  // real B rows; rest zero-filled

    // A fragment: lane group 0 holds K {0..15}, group 1 holds {16..31}
    int arow = m0 + lr; if (arow > M - 1) arow = M - 1;
    const bf16_t* ap = A + (long)arow * lda + half * 16;

    const bf16_t* bsrc = B + (long)n0 * ldb;
    const unsigned lds0 = (unsigned)(uintptr_t)(&Bs[0][0]);
    const unsigned lds1 = (unsigned)(uintptr_t)(&Bs[1][0]);
    const bool tdm_wave = (threadIdx.x < 32);

    const int nchunks = (K + KC - 1) / KC;
    int chunk0 = K > KC ? KC : K;

    if (tdm_wave) {
        tdm_load_tile(bsrc, lds0, chunk0, rows_rem, ldb);
        __builtin_amdgcn_s_wait_tensorcnt(0);
    }
    __syncthreads();

    v8f acc0 = {}; v8f acc1 = {}; v8f acc2 = {}; v8f acc3 = {};

    for (int c = 0; c < nchunks; ++c) {
        const int k0 = c * KC;
        int chunk = K - k0; if (chunk > KC) chunk = KC;

        if (c + 1 < nchunks && tdm_wave) {
            int nchunk = K - (k0 + KC); if (nchunk > KC) nchunk = KC;
            tdm_load_tile(bsrc + (k0 + KC), (c & 1) ? lds0 : lds1, nchunk, rows_rem, ldb);
        }

        const bf16_t* bs = &Bs[c & 1][0];

        if (chunk == KC) {
            // -------- hot path: full chunk, fully unrolled, constant strides ----
            v8bf alo = *(const v8bf*)(ap + k0);
            v8bf ahi = *(const v8bf*)(ap + k0 + 8);
#pragma unroll
            for (int kk = 0; kk < KC; kk += 32) {
                v8bf nlo = {}, nhi = {};
                if (kk + 32 < KC) {
                    nlo = *(const v8bf*)(ap + k0 + kk + 32);
                    nhi = *(const v8bf*)(ap + k0 + kk + 40);
                }
                v16bf af;
#pragma unroll
                for (int q = 0; q < 8; ++q) { af[q] = alo[q]; af[8 + q] = ahi[q]; }

                const int koff = kk + half * 16;
                v16bf b0 = *(const v16bf*)(bs + (lr +  0) * KC + koff);
                v16bf b1 = *(const v16bf*)(bs + (lr + 16) * KC + koff);
                v16bf b2 = *(const v16bf*)(bs + (lr + 32) * KC + koff);
                v16bf b3 = *(const v16bf*)(bs + (lr + 48) * KC + koff);

                acc0 = __builtin_amdgcn_wmma_f32_16x16x32_bf16(false, af, false, b0, (short)0, acc0, false, false);
                acc1 = __builtin_amdgcn_wmma_f32_16x16x32_bf16(false, af, false, b1, (short)0, acc1, false, false);
                acc2 = __builtin_amdgcn_wmma_f32_16x16x32_bf16(false, af, false, b2, (short)0, acc2, false, false);
                acc3 = __builtin_amdgcn_wmma_f32_16x16x32_bf16(false, af, false, b3, (short)0, acc3, false, false);

                alo = nlo; ahi = nhi;
            }
            if (k0 + KC < K) __builtin_prefetch(ap + k0 + KC, 0, 3);
        } else {
            // -------- tail path: partial chunk (K < KC multiples) ---------------
            for (int kk = 0; kk < chunk; kk += 32) {
                v8bf alo = *(const v8bf*)(ap + k0 + kk);
                v8bf ahi = *(const v8bf*)(ap + k0 + kk + 8);
                v16bf af;
#pragma unroll
                for (int q = 0; q < 8; ++q) { af[q] = alo[q]; af[8 + q] = ahi[q]; }

                const int koff = kk + half * 16;
                v16bf b0 = *(const v16bf*)(bs + (long)(lr +  0) * chunk + koff);
                v16bf b1 = *(const v16bf*)(bs + (long)(lr + 16) * chunk + koff);
                v16bf b2 = *(const v16bf*)(bs + (long)(lr + 32) * chunk + koff);
                v16bf b3 = *(const v16bf*)(bs + (long)(lr + 48) * chunk + koff);

                acc0 = __builtin_amdgcn_wmma_f32_16x16x32_bf16(false, af, false, b0, (short)0, acc0, false, false);
                acc1 = __builtin_amdgcn_wmma_f32_16x16x32_bf16(false, af, false, b1, (short)0, acc1, false, false);
                acc2 = __builtin_amdgcn_wmma_f32_16x16x32_bf16(false, af, false, b2, (short)0, acc2, false, false);
                acc3 = __builtin_amdgcn_wmma_f32_16x16x32_bf16(false, af, false, b3, (short)0, acc3, false, false);
            }
        }

        if (c + 1 < nchunks) {
            if (tdm_wave) __builtin_amdgcn_s_wait_tensorcnt(0);
            __syncthreads();
        }
    }

    // C/D layout: VGPR v, lanes 0-15: M=v, N=lane; lanes 16-31: M=8+v, N=lane-16
#pragma unroll
    for (int v = 0; v < 8; ++v) {
        const int rr = m0 + half * 8 + v;
        if (rr >= M) continue;
        float* crow = C + (long)rr * ldc;
        int cc;
        float val;
        cc = n0 + 0  + lr; if (cc < N) { val = aeff * acc0[v] + (bias ? bias[cc] : 0.f); crow[cc] = accum ? crow[cc] + val : val; }
        cc = n0 + 16 + lr; if (cc < N) { val = aeff * acc1[v] + (bias ? bias[cc] : 0.f); crow[cc] = accum ? crow[cc] + val : val; }
        cc = n0 + 32 + lr; if (cc < N) { val = aeff * acc2[v] + (bias ? bias[cc] : 0.f); crow[cc] = accum ? crow[cc] + val : val; }
        cc = n0 + 48 + lr; if (cc < N) { val = aeff * acc3[v] + (bias ? bias[cc] : 0.f); crow[cc] = accum ? crow[cc] + val : val; }
    }
}

// ===========================================================================
// Elementwise / reduction helper kernels
// ===========================================================================
__global__ void k_cast_bf16(const float* __restrict__ s, bf16_t* __restrict__ d, long n)
{
    long i = blockIdx.x * (long)blockDim.x + threadIdx.x;
    if (i < n) d[i] = (bf16_t)s[i];
}

__global__ void k_fill_f32(float* __restrict__ p, float v, long n)
{
    long i = blockIdx.x * (long)blockDim.x + threadIdx.x;
    if (i < n) p[i] = v;
}

// score[b][n] = exp(x[b,n,:] . score_w + score_b)
__global__ void k_score(const float* __restrict__ x, const float* __restrict__ sw,
                        const float* __restrict__ sb, float* __restrict__ score)
{
    int i = blockIdx.x * blockDim.x + threadIdx.x;
    if (i >= BB * NN) return;
    const float* row = x + (long)i * CC;
    float s = 0.f;
    for (int c = 0; c < CC; ++c) s += row[c] * sw[c];
    score[i] = expf(s + sb[0]);
}

// mean[b][c] = mean over tokens
__global__ void k_colmean(const float* __restrict__ x, float* __restrict__ mean)
{
    int i = blockIdx.x * blockDim.x + threadIdx.x;
    if (i >= BB * CC) return;
    int b = i / CC, c = i % CC;
    float s = 0.f;
    for (int n2 = 0; n2 < NN; ++n2) s += x[((long)b * NN + n2) * CC + c];
    mean[i] = s / (float)NN;
}

// gating: weights = sigmoid(mean @ route_w^T + route_b); top-2 experts
__global__ void k_route(const float* __restrict__ mean, const float* __restrict__ rw,
                        const float* __restrict__ rb, float* __restrict__ weights,
                        int* __restrict__ topk)
{
    const int b = blockIdx.x;
    const int wv = threadIdx.x >> 5, lane = threadIdx.x & 31;
    __shared__ float wsh[EE];
    if (wv < EE) {
        float s = 0.f;
        for (int c = lane; c < CC; c += 32) s += mean[b * CC + c] * rw[wv * CC + c];
        for (int o = 16; o; o >>= 1) s += __shfl_down(s, o, 32);
        if (lane == 0) wsh[wv] = 1.f / (1.f + expf(-(s + rb[wv])));
    }
    __syncthreads();
    if (threadIdx.x == 0) {
        for (int e = 0; e < EE; ++e) weights[b * EE + e] = wsh[e];
        int i0 = 0;
        for (int e = 1; e < EE; ++e) if (wsh[e] > wsh[i0]) i0 = e;
        int i1 = -1;
        for (int e = 0; e < EE; ++e) { if (e == i0) continue; if (i1 < 0 || wsh[e] > wsh[i1]) i1 = e; }
        topk[b * 2 + 0] = i0;
        topk[b * 2 + 1] = i1;
    }
}

// Single-block bitonic argsort over global scratch (handles S up to 16384).
__global__ void k_bitonic_argsort(const float* __restrict__ keys, int n, int S, int ascending,
                                  float* __restrict__ kb, int* __restrict__ ib,
                                  float* __restrict__ okeys, int* __restrict__ oidx)
{
    const float pad = ascending ? FINF : -FINF;
    for (int i = threadIdx.x; i < S; i += blockDim.x) {
        kb[i] = (i < n) ? keys[i] : pad;
        ib[i] = i;
    }
    __syncthreads();
    for (int k = 2; k <= S; k <<= 1) {
        for (int j = k >> 1; j > 0; j >>= 1) {
            for (int i = threadIdx.x; i < S; i += blockDim.x) {
                int ixj = i ^ j;
                if (ixj > i) {
                    bool dirAsc = ((i & k) == 0) ? (ascending != 0) : (ascending == 0);
                    float a = kb[i], b2 = kb[ixj];
                    bool sw = dirAsc ? (a > b2) : (a < b2);
                    if (sw) { kb[i] = b2; kb[ixj] = a; int t = ib[i]; ib[i] = ib[ixj]; ib[ixj] = t; }
                }
            }
            __syncthreads();
        }
    }
    for (int i = threadIdx.x; i < n; i += blockDim.x) { okeys[i] = kb[i]; oidx[i] = ib[i]; }
}

// xs[b][n][c] = x[b][order[b][n]][c]   (fp32 + bf16)
__global__ void k_gather(const float* __restrict__ x, const int* __restrict__ order,
                         float* __restrict__ xs, bf16_t* __restrict__ xsb)
{
    long i = blockIdx.x * (long)blockDim.x + threadIdx.x;
    if (i >= (long)BB * NN * CC) return;
    int c = (int)(i % CC);
    long t = i / CC;
    int n2 = (int)(t % NN);
    int b = (int)(t / NN);
    float v = x[((long)b * NN + order[b * NN + n2]) * CC + c];
    xs[i] = v;
    xsb[i] = (bf16_t)v;
}

// agg_nums per batch: region dot with rscale_w, softmax over 3, clip(320*p,16,320)
__global__ void k_agg(const float* __restrict__ ss, const float* __restrict__ rsw,
                      const float* __restrict__ rsb, int* __restrict__ agg)
{
    const int b = blockIdx.x;
    const int wv = threadIdx.x >> 5, lane = threadIdx.x & 31;
    __shared__ float sc[3];
    if (wv < 3) {
        float s = 0.f;
        for (int t = lane; t < NR3; t += 32) s += ss[b * NN + wv * NR3 + t] * rsw[wv * NR3 + t];
        for (int o = 16; o; o >>= 1) s += __shfl_down(s, o, 32);
        if (lane == 0) sc[wv] = s + rsb[wv];
    }
    __syncthreads();
    if (threadIdx.x == 0) {
        float m = fmaxf(sc[0], fmaxf(sc[1], sc[2]));
        float e0 = expf(sc[0] - m), e1 = expf(sc[1] - m), e2 = expf(sc[2] - m);
        float tot = e0 + e1 + e2;
        float es[3] = { e0, e1, e2 };
        for (int r = 0; r < 3; ++r) {
            float v = 320.f * es[r] / tot;
            v = fminf(fmaxf(v, 16.f), 320.f);
            agg[b * 3 + r] = (int)v;
        }
    }
}

// sq[i] = |toks[i]|^2
__global__ void k_sqnorm(const float* __restrict__ toks, float* __restrict__ sq, int n)
{
    int i = blockIdx.x * blockDim.x + threadIdx.x;
    if (i >= n) return;
    const float* r = toks + (long)i * CC;
    float s = 0.f;
    for (int c = 0; c < CC; ++c) s += r[c] * r[c];
    sq[i] = s;
}

// Gram -> distance in place; track global max (float atomicMax via int bits, d>=0)
__global__ void k_dist(float* __restrict__ G, const float* __restrict__ sq,
                       float* __restrict__ dmax, int n)
{
    const long total = (long)n * n;
    const float rs = rsqrtf((float)CC);
    float lmax = 0.f;
    for (long t = blockIdx.x * (long)blockDim.x + threadIdx.x; t < total;
         t += (long)gridDim.x * blockDim.x) {
        int i = (int)(t / n), j = (int)(t % n);
        float d2 = sq[i] + sq[j] - 2.f * G[t];
        d2 = fmaxf(d2, 0.f);
        float d = sqrtf(d2) * rs;
        G[t] = d;
        lmax = fmaxf(lmax, d);
    }
    __shared__ float red[32];
    const int lane = threadIdx.x & 31, wv = threadIdx.x >> 5;
    for (int o = 16; o; o >>= 1) lmax = fmaxf(lmax, __shfl_down(lmax, o, 32));
    if (lane == 0) red[wv] = lmax;
    __syncthreads();
    if (wv == 0) {
        float v = (lane < (int)(blockDim.x >> 5)) ? red[lane] : 0.f;
        for (int o = 16; o; o >>= 1) v = fmaxf(v, __shfl_down(v, o, 32));
        if (lane == 0) atomicMax((int*)dmax, __float_as_int(v));
    }
}

// k-NN density (k = floor(sqrt(num)) <= 17) with deterministic hash tiebreak
__global__ void k_knn_density(const float* __restrict__ D, const int* __restrict__ numptr,
                              float* __restrict__ density, int n, unsigned seed)
{
    int i = blockIdx.x * blockDim.x + threadIdx.x;
    if (i >= n) return;
    int num = numptr[0];
    int kk = (int)sqrtf((float)num);
    if (kk < 1) kk = 1;
    if (kk > 17) kk = 17;
    float nb[18];
    for (int t = 0; t < kk; ++t) nb[t] = FINF;
    const float* row = D + (long)i * n;
    for (int j = 0; j < n; ++j) {
        float d = row[j];
        if (d < nb[kk - 1]) {
            int p = kk - 1;
            while (p > 0 && nb[p - 1] > d) { nb[p] = nb[p - 1]; --p; }
            nb[p] = d;
        }
    }
    float s = 0.f;
    for (int t = 0; t < kk; ++t) s += nb[t] * nb[t];
    float dens = expf(-s / (float)kk);
    unsigned h = (unsigned)i * 2654435761u + seed * 2246822519u;
    h ^= h >> 13; h *= 2654435761u; h ^= h >> 16;
    density[i] = dens + (float)(h & 0xFFFFFFu) * (1.0f / 16777216.0f) * 1e-6f;
}

// dmin over higher-density points; cluster score = dmin * density
__global__ void k_dmin_score(const float* __restrict__ D, const float* __restrict__ dens,
                             const float* __restrict__ dmax, float* __restrict__ sc, int n)
{
    int i = blockIdx.x * blockDim.x + threadIdx.x;
    if (i >= n) return;
    float di = dens[i], dm = dmax[0];
    const float* row = D + (long)i * n;
    float mn = dm;
    for (int j = 0; j < n; ++j) {
        float v = (dens[j] > di) ? row[j] : dm;
        mn = fminf(mn, v);
    }
    sc[i] = mn * di;
}

// assign each token to nearest of the `num` cluster centers
__global__ void k_assign(const float* __restrict__ D, const int* __restrict__ idown,
                         const int* __restrict__ numptr, int* __restrict__ idxc, int n)
{
    int j = blockIdx.x * blockDim.x + threadIdx.x;
    if (j >= n) return;
    int num = numptr[0];
    float best = FINF; int bi = 0;
    for (int c = 0; c < num; ++c) {
        float d = D[(long)idown[c] * n + j];
        if (d < best) { best = d; bi = c; }
    }
    idxc[j] = bi;
}

__global__ void k_assign_fix(const int* __restrict__ idown, const int* __restrict__ numptr,
                             int* __restrict__ idxc)
{
    int c = blockIdx.x * blockDim.x + threadIdx.x;
    if (c < numptr[0]) idxc[idown[c]] = c;
}

__global__ void k_allw(const int* __restrict__ idxc, const float* __restrict__ tw,
                       float* __restrict__ allw, int n)
{
    int j = blockIdx.x * blockDim.x + threadIdx.x;
    if (j < n) atomicAdd(&allw[idxc[j]], tw[j]);
}

// merged[c] += toks[j] * tw[j] / allw[c]   (one block per token)
__global__ void k_merge(const float* __restrict__ toks, const float* __restrict__ tw,
                        const int* __restrict__ idxc, const float* __restrict__ allw,
                        float* __restrict__ dst)
{
    const int j = blockIdx.x;
    const int c = idxc[j];
    const float w = tw[j] / allw[c];
    const float* srow = toks + (long)j * CC;
    float* drow = dst + (long)c * CC;
    for (int ch = threadIdx.x; ch < CC; ch += blockDim.x)
        atomicAdd(&drow[ch], srow[ch] * w);
}

// validity mask of padded merged-token slots
__global__ void k_valid(const int* __restrict__ agg, int* __restrict__ valid)
{
    int i = blockIdx.x * blockDim.x + threadIdx.x;
    if (i >= BB * MP) return;
    int b = i / MP, m = i % MP;
    int r = m / MAXAGG, s2 = m % MAXAGG;
    valid[i] = (s2 < agg[b * 3 + r]) ? 1 : 0;
}

// qfull [B][N][C] fp32 -> qh [B][H][N][DP] bf16 (zero-padded head dim)
__global__ void k_pack_q(const float* __restrict__ qfull, bf16_t* __restrict__ qh)
{
    long i = blockIdx.x * (long)blockDim.x + threadIdx.x;
    const long total = (long)BB * HH * NN * DP;
    if (i >= total) return;
    int p = (int)(i % DP); long t = i / DP;
    int n2 = (int)(t % NN); t /= NN;
    int h = (int)(t % HH); int b = (int)(t / HH);
    float v = (p < DD) ? qfull[((long)b * NN + n2) * CC + h * DD + p] : 0.f;
    qh[i] = (bf16_t)v;
}

// kvfull [MP][2C] -> kh [H][MP][DP] bf16 (padded)
__global__ void k_pack_kh(const float* __restrict__ kvfull, bf16_t* __restrict__ kh)
{
    long i = blockIdx.x * (long)blockDim.x + threadIdx.x;
    const long total = (long)HH * MP * DP;
    if (i >= total) return;
    int p = (int)(i % DP); long t = i / DP;
    int m = (int)(t % MP); int h = (int)(t / MP);
    float v = (p < DD) ? kvfull[(long)m * (2 * CC) + h * DD + p] : 0.f;
    kh[i] = (bf16_t)v;
}

// kvfull -> vT [H][DD][MP] bf16
__global__ void k_pack_vt(const float* __restrict__ kvfull, bf16_t* __restrict__ vt)
{
    long i = blockIdx.x * (long)blockDim.x + threadIdx.x;
    const long total = (long)HH * DD * MP;
    if (i >= total) return;
    int m = (int)(i % MP); long t = i / MP;
    int dd = (int)(t % DD); int h = (int)(t / DD);
    vt[i] = (bf16_t)kvfull[(long)m * (2 * CC) + CC + h * DD + dd];
}

// masked row softmax over MP columns; writes bf16 probabilities
__global__ void k_softmax_mask(float* __restrict__ S, const int* __restrict__ valid,
                               bf16_t* __restrict__ P, int cols, float scale)
{
    __shared__ float red[8];
    __shared__ float red2[8];
    const int row = blockIdx.x;
    float* srow = S + (long)row * cols;
    bf16_t* prow = P + (long)row * cols;
    const int tid = threadIdx.x, lane = tid & 31, wv = tid >> 5, nw = blockDim.x >> 5;

    float mx = -FINF;
    for (int c = tid; c < cols; c += blockDim.x) {
        float v = valid[c] ? srow[c] * scale : -FINF;
        srow[c] = v;
        mx = fmaxf(mx, v);
    }
    for (int o = 16; o; o >>= 1) mx = fmaxf(mx, __shfl_down(mx, o, 32));
    if (lane == 0) red[wv] = mx;
    __syncthreads();
    if (tid == 0) { float m = red[0]; for (int w2 = 1; w2 < nw; ++w2) m = fmaxf(m, red[w2]); red[0] = m; }
    __syncthreads();
    mx = red[0];

    float sum = 0.f;
    for (int c = tid; c < cols; c += blockDim.x) {
        float e = expf(srow[c] - mx);
        srow[c] = e;
        sum += e;
    }
    for (int o = 16; o; o >>= 1) sum += __shfl_down(sum, o, 32);
    if (lane == 0) red2[wv] = sum;
    __syncthreads();
    if (tid == 0) { float s2 = 0.f; for (int w2 = 0; w2 < nw; ++w2) s2 += red2[w2]; red2[0] = s2; }
    __syncthreads();
    const float rsum = 1.f / red2[0];
    for (int c = tid; c < cols; c += blockDim.x) prow[c] = (bf16_t)(srow[c] * rsum);
}

// ===========================================================================
// Host orchestration
// ===========================================================================
static inline void gemm_nt(hipStream_t s, const bf16_t* A, long lda, const bf16_t* B, long ldb,
                           float* C, long ldc, int M, int N, int K,
                           float alpha = 1.f, const float* avec = nullptr, const int* asel = nullptr,
                           const int* bsel = nullptr, long bstride = 0,
                           const float* bias = nullptr, int accum = 0)
{
    dim3 g((N + 63) / 64, (M + 127) / 128);
    k_wmma_gemm_nt<<<g, 256, 0, s>>>(A, lda, B, ldb, C, ldc, M, N, K,
                                     alpha, avec, asel, bsel, bstride, bias, accum);
}

static inline dim3 g1(long n, int bs) { return dim3((unsigned)((n + bs - 1) / bs)); }

extern "C" void kernel_launch(void* const* d_in, const int* in_sizes, int n_in,
                              void* d_out, int out_size, void* d_ws, size_t ws_size,
                              hipStream_t stream)
{
    (void)in_sizes; (void)n_in; (void)out_size; (void)ws_size;
    const float* x        = (const float*)d_in[0];
    const float* q_w      = (const float*)d_in[1];
    const float* kv_w     = (const float*)d_in[2];
    const float* route_w  = (const float*)d_in[3];
    const float* route_b  = (const float*)d_in[4];
    const float* score_w  = (const float*)d_in[5];
    const float* score_b  = (const float*)d_in[6];
    const float* rscale_w = (const float*)d_in[7];
    const float* rscale_b = (const float*)d_in[8];
    const float* proj_w   = (const float*)d_in[9];
    const float* proj_b   = (const float*)d_in[10];
    float* out = (float*)d_out;

    char* base = (char*)d_ws;
    size_t off = 0;
    auto alloc = [&](size_t bytes) -> char* {
        off = (off + 255) & ~(size_t)255;
        char* r = base + off;
        off += bytes;
        return r;
    };

    bf16_t* x_bf    = (bf16_t*)alloc((size_t)BB * NN * CC * 2);
    float*  score   = (float*) alloc((size_t)BB * NN * 4);
    float*  mean    = (float*) alloc((size_t)BB * CC * 4);
    float*  weights = (float*) alloc((size_t)BB * EE * 4);
    int*    topk    = (int*)   alloc((size_t)BB * 2 * 4);
    float*  kbuf    = (float*) alloc((size_t)SORT_S * 4);
    int*    ibuf    = (int*)   alloc((size_t)SORT_S * 4);
    int*    order   = (int*)   alloc((size_t)BB * NN * 4);
    float*  ss      = (float*) alloc((size_t)BB * NN * 4);
    float*  xs_f    = (float*) alloc((size_t)BB * NN * CC * 4);
    bf16_t* xs_bf   = (bf16_t*)alloc((size_t)BB * NN * CC * 2);
    int*    agg     = (int*)   alloc((size_t)BB * 3 * 4);
    bf16_t* qwb     = (bf16_t*)alloc((size_t)CC * CC * 2);
    bf16_t* kvwb    = (bf16_t*)alloc((size_t)EE * 2 * CC * CC * 2);
    bf16_t* pwb     = (bf16_t*)alloc((size_t)CC * CC * 2);
    float*  dist    = (float*) alloc((size_t)NR3 * NR3 * 4);     // also reused as attn logits S
    float*  sq      = (float*) alloc((size_t)NR3 * 4);
    float*  density = (float*) alloc((size_t)NR3 * 4);
    float*  scorec  = (float*) alloc((size_t)NR3 * 4);
    float*  dmaxb   = (float*) alloc(4);
    float*  tkeys   = (float*) alloc((size_t)TOPK_S * 4);
    int*    idown   = (int*)   alloc((size_t)TOPK_S * 4);
    int*    idxc    = (int*)   alloc((size_t)NR3 * 4);
    float*  allw    = (float*) alloc((size_t)MAXAGG * 4);
    float*  aggX    = (float*) alloc((size_t)BB * MP * CC * 4);
    bf16_t* aggXb   = (bf16_t*)alloc((size_t)BB * MP * CC * 2);
    int*    valid   = (int*)   alloc((size_t)BB * MP * 4);
    float*  qfull   = (float*) alloc((size_t)BB * NN * CC * 4); // reused as outfull
    bf16_t* qh      = (bf16_t*)alloc((size_t)BB * HH * NN * DP * 2);
    float*  kvfull  = (float*) alloc((size_t)MP * 2 * CC * 4);
    bf16_t* kh      = (bf16_t*)alloc((size_t)HH * MP * DP * 2);
    bf16_t* vt      = (bf16_t*)alloc((size_t)HH * DD * MP * 2);
    bf16_t* P       = (bf16_t*)alloc((size_t)NN * MP * 2);
    float*  Sbuf    = dist;          // alias: clustering done before attention
    float*  outfull = qfull;         // alias: qfull dead after k_pack_q
    bf16_t* out_bf  = xs_bf;         // alias: xs_bf dead after clustering

    // ---- Phase 0: casts, scores, gating ----
    k_cast_bf16<<<g1((long)BB * NN * CC, 256), 256, 0, stream>>>(x, x_bf, (long)BB * NN * CC);
    k_cast_bf16<<<g1((long)CC * CC, 256), 256, 0, stream>>>(q_w, qwb, (long)CC * CC);
    k_cast_bf16<<<g1((long)EE * 2 * CC * CC, 256), 256, 0, stream>>>(kv_w, kvwb, (long)EE * 2 * CC * CC);
    k_cast_bf16<<<g1((long)CC * CC, 256), 256, 0, stream>>>(proj_w, pwb, (long)CC * CC);
    k_score<<<g1((long)BB * NN, 256), 256, 0, stream>>>(x, score_w, score_b, score);
    k_colmean<<<g1((long)BB * CC, 256), 256, 0, stream>>>(x, mean);
    k_route<<<BB, 128, 0, stream>>>(mean, route_w, route_b, weights, topk);

    // ---- Phase 1: per-batch ascending argsort of scores, gather sorted tokens ----
    for (int b = 0; b < BB; ++b)
        k_bitonic_argsort<<<1, 1024, 0, stream>>>(score + (long)b * NN, NN, SORT_S, 1,
                                                  kbuf, ibuf, ss + (long)b * NN, order + (long)b * NN);
    k_gather<<<g1((long)BB * NN * CC, 256), 256, 0, stream>>>(x, order, xs_f, xs_bf);
    k_agg<<<BB, 96, 0, stream>>>(ss, rscale_w, rscale_b, agg);

    // ---- Phase 2: density-peak clustering per (batch, region) ----
    k_fill_f32<<<g1((long)BB * MP * CC, 256), 256, 0, stream>>>(aggX, 0.f, (long)BB * MP * CC);
    for (int b = 0; b < BB; ++b) {
        for (int r = 0; r < 3; ++r) {
            const float*  toks_f = xs_f  + ((long)b * NN + r * NR3) * CC;
            const bf16_t* toks_b = xs_bf + ((long)b * NN + r * NR3) * CC;
            const float*  tw     = ss + (long)b * NN + r * NR3;
            const int*    numptr = agg + b * 3 + r;
            float* dstX = aggX + ((long)b * MP + r * MAXAGG) * CC;

            k_sqnorm<<<g1(NR3, 256), 256, 0, stream>>>(toks_f, sq, NR3);
            gemm_nt(stream, toks_b, CC, toks_b, CC, dist, NR3, NR3, NR3, CC);
            k_fill_f32<<<1, 32, 0, stream>>>(dmaxb, 0.f, 1);
            k_dist<<<1024, 256, 0, stream>>>(dist, sq, dmaxb, NR3);
            k_knn_density<<<g1(NR3, 256), 256, 0, stream>>>(dist, numptr, density, NR3,
                                                            (unsigned)(b * 3 + r + 1));
            k_dmin_score<<<g1(NR3, 256), 256, 0, stream>>>(dist, density, dmaxb, scorec, NR3);
            k_bitonic_argsort<<<1, 1024, 0, stream>>>(scorec, NR3, TOPK_S, 0, kbuf, ibuf, tkeys, idown);
            k_assign<<<g1(NR3, 256), 256, 0, stream>>>(dist, idown, numptr, idxc, NR3);
            k_assign_fix<<<g1(MAXAGG, 256), 256, 0, stream>>>(idown, numptr, idxc);
            k_fill_f32<<<g1(MAXAGG, 256), 256, 0, stream>>>(allw, 1e-6f, MAXAGG);
            k_allw<<<g1(NR3, 256), 256, 0, stream>>>(idxc, tw, allw, NR3);
            k_merge<<<NR3, 128, 0, stream>>>(toks_f, tw, idxc, allw, dstX);
        }
    }
    k_cast_bf16<<<g1((long)BB * MP * CC, 256), 256, 0, stream>>>(aggX, aggXb, (long)BB * MP * CC);
    k_valid<<<g1((long)BB * MP, 256), 256, 0, stream>>>(agg, valid);

    // ---- Phase 3: Q projection (one big WMMA GEMM over all batches) ----
    gemm_nt(stream, x_bf, CC, qwb, CC, qfull, CC, BB * NN, CC, CC);
    k_pack_q<<<g1((long)BB * HH * NN * DP, 256), 256, 0, stream>>>(qfull, qh);

    // ---- Phase 4: expert attention (outfull aliases qfull; zero AFTER pack) ----
    k_fill_f32<<<g1((long)BB * NN * CC, 256), 256, 0, stream>>>(outfull, 0.f, (long)BB * NN * CC);
    const float attn_scale = rsqrtf((float)DD);
    for (int b = 0; b < BB; ++b) {
        for (int j = 0; j < 2; ++j) {
            const int* esel = topk + b * 2 + j;
            // kv = agg @ kv_w[e]^T  (expert index resolved on device)
            gemm_nt(stream, aggXb + (long)b * MP * CC, CC, kvwb, CC, kvfull, 2 * CC,
                    MP, 2 * CC, CC, 1.f, nullptr, nullptr, esel, (long)2 * CC * CC);
            k_pack_kh<<<g1((long)HH * MP * DP, 256), 256, 0, stream>>>(kvfull, kh);
            k_pack_vt<<<g1((long)HH * DD * MP, 256), 256, 0, stream>>>(kvfull, vt);
            for (int h = 0; h < HH; ++h) {
                // logits = Qh @ Kh^T
                gemm_nt(stream, qh + ((long)(b * HH + h)) * NN * DP, DP,
                        kh + (long)h * MP * DP, DP, Sbuf, MP, NN, MP, DP);
                k_softmax_mask<<<NN, 256, 0, stream>>>(Sbuf, valid + (long)b * MP, P, MP, attn_scale);
                // out[:, h*48:(h+1)*48] += w_e * P @ V
                gemm_nt(stream, P, MP, vt + (long)h * DD * MP, MP,
                        outfull + (long)b * NN * CC + h * DD, CC, NN, DD, MP,
                        1.f, weights + b * EE, esel, nullptr, 0, nullptr, 1);
            }
        }
    }

    // ---- Phase 5: output projection with bias ----
    k_cast_bf16<<<g1((long)BB * NN * CC, 256), 256, 0, stream>>>(outfull, out_bf, (long)BB * NN * CC);
    gemm_nt(stream, out_bf, CC, pwb, CC, out, CC, BB * NN, CC, CC,
            1.f, nullptr, nullptr, nullptr, 0, proj_b, 0);
}
